// GRUNet_53961969107392
// MI455X (gfx1250) — matrix operand order repaired and checked by reading, not compiled, not run on hardware
//
#include <hip/hip_runtime.h>
#include <hip/hip_bf16.h>

// ---------------------------------------------------------------------------
// 2-layer GRU, persistent-kernel, bf16 WMMA + LDS-resident weights (gfx1250)
//   B=256  T=1024  H=512  gates=3H=1536
// Each of the 64 WGs stages its weight strips into LDS ONCE (weights are
// timestep-invariant), then loops t:
//   phase A: GH1 = h1b*Wc1^T (WG 0-31) | GH2 = h2b*Wh2^T (WG 32-63)
//   phase B: h1 gate math (waves 0-255) | y[:,t-1] readout (waves 256-511)
//   phase C: GI2 = h1b*Wi2^T (WG pairs share a strip, split by M)
//   phase D: h2 gate math (all threads)
// ---------------------------------------------------------------------------

#define Bsz   256
#define Tsz   1024
#define Hsz   512
#define G3    1536
#define NWG   64
#define TPB   256                 // 8 waves of 32
#define SCOLS 48                  // weight-strip columns per WG
#define SMEM_BYTES (2u * SCOLS * Hsz * 2u)   // two bf16 strips = 96 KB

typedef __attribute__((ext_vector_type(16))) __bf16 v16bf;
typedef __attribute__((ext_vector_type(8)))  __bf16 v8bf;
typedef __attribute__((ext_vector_type(8)))  float  v8f;

__device__ __forceinline__ unsigned short f2bf(float f) {
    unsigned int u = __float_as_uint(f);
    u += 0x7FFFu + ((u >> 16) & 1u);          // round-to-nearest-even
    return (unsigned short)(u >> 16);
}
__device__ __forceinline__ float sigm(float x) { return 1.0f / (1.0f + __expf(-x)); }

// ---- WMMA fragment loaders (ISA 7.12.2 VGPR layouts, wave32) ---------------
// A (16x32, 16-bit): lane m (0-15): K = {0..7,16..23}; lane m+16: K = {8..15,24..31}
__device__ __forceinline__ v16bf load_fragA(const unsigned short* __restrict__ A,
                                            int rowBase, int kOff, int lane) {
    int m  = lane & 15;
    int kb = (lane >> 4) << 3;
    const unsigned short* r = A + (size_t)(rowBase + m) * Hsz + kOff + kb;
    v8bf lo = *reinterpret_cast<const v8bf*>(r);        // K = kb .. kb+7
    v8bf hi = *reinterpret_cast<const v8bf*>(r + 16);   // K = kb+16 .. kb+23
    return __builtin_shufflevector(lo, hi, 0,1,2,3,4,5,6,7,8,9,10,11,12,13,14,15);
}
// B (32x16, 16-bit) out of LDS: lane n (0-15): K={0..7,16..23}; lane n+16: K={8..15,24..31}.
// LDS strip is row-major (strip-local col, k), ld = Hsz.
__device__ __forceinline__ v16bf load_fragB_lds(const unsigned short* Wlds,
                                                int colLocal, int kOff, int lane) {
    int n  = lane & 15;
    int k0 = kOff + ((lane >> 4) << 3);
    const unsigned short* r = Wlds + (colLocal + n) * Hsz + k0;
    v8bf lo = *reinterpret_cast<const v8bf*>(r);
    v8bf hi = *reinterpret_cast<const v8bf*>(r + 16);
    return __builtin_shufflevector(lo, hi, 0,1,2,3,4,5,6,7,8,9,10,11,12,13,14,15);
}

// MT x 3 macro-tile GEMM strip vs LDS-resident weights, K = 512
template <int MT>
__device__ __forceinline__ void gemm_strip_lds(const unsigned short* __restrict__ Amat,
                                               const unsigned short* Wlds,
                                               float* __restrict__ G,
                                               int mbase, int nbGlobal, int lane) {
    v8f acc[MT][3];
#pragma unroll
    for (int i = 0; i < MT; ++i)
#pragma unroll
        for (int j = 0; j < 3; ++j) acc[i][j] = (v8f){0,0,0,0,0,0,0,0};

    for (int kk = 0; kk < Hsz; kk += 32) {
        v16bf a[MT];
#pragma unroll
        for (int i = 0; i < MT; ++i) a[i] = load_fragA(Amat, mbase + 16 * i, kk, lane);
#pragma unroll
        for (int j = 0; j < 3; ++j) {
            v16bf b = load_fragB_lds(Wlds, 16 * j, kk, lane);
#pragma unroll
            for (int i = 0; i < MT; ++i)
                acc[i][j] = __builtin_amdgcn_wmma_f32_16x16x32_bf16(
                    false, a[i], false, b, (short)0, acc[i][j], false, false);
        }
    }
    // D layout: lane L, vgpr r -> row = r + 8*(L>>4), col = L&15
    int nn = lane & 15, half = lane >> 4;
#pragma unroll
    for (int i = 0; i < MT; ++i)
#pragma unroll
        for (int j = 0; j < 3; ++j)
#pragma unroll
            for (int r = 0; r < 8; ++r)
                G[(size_t)(mbase + 16 * i + 8 * half + r) * G3 + nbGlobal + 16 * j + nn] =
                    acc[i][j][r];
}

// copy one 48x512 bf16 strip (49152 B) global -> LDS, whole WG cooperating
__device__ __forceinline__ void stage_strip(unsigned short* dstLds,
                                            const unsigned short* __restrict__ src,
                                            int tid) {
    const uint4* s = (const uint4*)src;
    uint4*       d = (uint4*)dstLds;
#pragma unroll
    for (int i = 0; i < (SCOLS * Hsz / 8) / TPB; ++i)     // 3072 / 256 = 12
        d[tid + i * TPB] = s[tid + i * TPB];
}

// ---- device-wide sense barrier (all NWG blocks resident) -------------------
__device__ __forceinline__ void gridbar(int* ctrl) {
    __threadfence();
    __syncthreads();
    if (threadIdx.x == 0) {
        int gen  = __hip_atomic_load(ctrl + 1, __ATOMIC_RELAXED, __HIP_MEMORY_SCOPE_AGENT);
        int prev = __hip_atomic_fetch_add(ctrl, 1, __ATOMIC_ACQ_REL, __HIP_MEMORY_SCOPE_AGENT);
        if (prev == NWG - 1) {
            __hip_atomic_store(ctrl, 0, __ATOMIC_RELAXED, __HIP_MEMORY_SCOPE_AGENT);
            __hip_atomic_fetch_add(ctrl + 1, 1, __ATOMIC_RELEASE, __HIP_MEMORY_SCOPE_AGENT);
        } else {
            while (__hip_atomic_load(ctrl + 1, __ATOMIC_ACQUIRE, __HIP_MEMORY_SCOPE_AGENT) == gen)
                __builtin_amdgcn_s_sleep(2);
        }
    }
    __syncthreads();
}

// ---- prep: fp32 weights -> bf16, zero state + barrier ----------------------
__global__ void gru_prep(const float* __restrict__ Whh1, const float* __restrict__ Wih2,
                         const float* __restrict__ Whh2,
                         unsigned short* wc1, unsigned short* wi2, unsigned short* wh2,
                         float* h1, float* h2, unsigned short* h1b, unsigned short* h2b,
                         int* ctrl) {
    int i      = blockIdx.x * blockDim.x + threadIdx.x;
    int stride = gridDim.x * blockDim.x;
    const int NW = G3 * Hsz;
    for (int k = i; k < NW; k += stride) {
        wc1[k] = f2bf(Whh1[k]);
        wi2[k] = f2bf(Wih2[k]);
        wh2[k] = f2bf(Whh2[k]);
    }
    const int NH = Bsz * Hsz;
    for (int k = i; k < NH; k += stride) {
        h1[k] = 0.f; h2[k] = 0.f; h1b[k] = 0; h2b[k] = 0;
    }
    if (i < 2) ctrl[i] = 0;
}

// ---- persistent GRU kernel -------------------------------------------------
__global__ __launch_bounds__(TPB, 1) void gru_persistent(
    const float* __restrict__ x,        // (B,T)
    const float* __restrict__ Wih1,     // (1536,)  since I==1
    const float* __restrict__ bih1, const float* __restrict__ bhh1,
    const float* __restrict__ bih2, const float* __restrict__ bhh2,
    const float* __restrict__ Wout,     // (512,)
    const float* __restrict__ bout,     // (1,)
    float* __restrict__ y,              // (B,T)
    const unsigned short* __restrict__ wc1, const unsigned short* __restrict__ wi2,
    const unsigned short* __restrict__ wh2,
    float* __restrict__ h1, float* __restrict__ h2,
    unsigned short* __restrict__ h1b, unsigned short* __restrict__ h2b,
    float* __restrict__ GH1, float* __restrict__ GI2, float* __restrict__ GH2,
    int* ctrl) {
    const int tid  = threadIdx.x;
    const int lane = tid & 31;
    const int wiw  = tid >> 5;                               // wave in WG, 0..7
    const int wave = blockIdx.x * (TPB / 32) + wiw;          // 0..511
    const int gtid = blockIdx.x * TPB + tid;                 // 0..16383

    // ---- one-time LDS staging of this WG's weight strips (t-invariant) ----
    extern __shared__ unsigned short smem_u16[];
    unsigned short* sW  = smem_u16;                 // phase A strip: Wc1 or Wh2
    unsigned short* sWi = smem_u16 + SCOLS * Hsz;   // phase C strip: Wi2
    {
        int g  = blockIdx.x >> 5;                   // 0: GH1 strip, 1: GH2 strip
        int sn = blockIdx.x & 31;
        stage_strip(sW, (g ? wh2 : wc1) + (size_t)(SCOLS * sn) * Hsz, tid);
        stage_strip(sWi, wi2 + (size_t)(SCOLS * (blockIdx.x >> 1)) * Hsz, tid);
    }
    __syncthreads();

    for (int t = 0; t < Tsz; ++t) {
        // ---- phase A: WG 0-31 -> GH1, WG 32-63 -> GH2; wave = 2x3 macro-tile
        {
            int mb = wiw * 32;                      // 8 waves cover 16 mtiles
            int nb = (blockIdx.x & 31) * SCOLS;
            if ((blockIdx.x >> 5) == 0) gemm_strip_lds<2>(h1b, sW, GH1, mb, nb, lane);
            else                        gemm_strip_lds<2>(h2b, sW, GH2, mb, nb, lane);
        }
        gridbar(ctrl);

        // ---- phase B: h1 gate update | readout y[:,t-1]
        if (wave < 256) {
            int th = wave * 32 + lane;              // 0..8191, 16 elems each
#pragma unroll 4
            for (int s = 0; s < 16; ++s) {
                int e = th + s * 8192;
                int b = e >> 9, j = e & 511;
                float xb = x[b * Tsz + t];
                float gr = xb * Wih1[j]        + bih1[j]        + GH1[(size_t)b * G3 + j]        + bhh1[j];
                float gz = xb * Wih1[j + 512]  + bih1[j + 512]  + GH1[(size_t)b * G3 + j + 512]  + bhh1[j + 512];
                float gn = xb * Wih1[j + 1024] + bih1[j + 1024];
                float hn = GH1[(size_t)b * G3 + j + 1024] + bhh1[j + 1024];
                float r = sigm(gr), z = sigm(gz);
                float n  = tanhf(gn + r * hn);
                float hv = (1.0f - z) * n + z * h1[e];
                h1[e]  = hv;
                h1b[e] = f2bf(hv);
            }
        } else if (t > 0) {
            int b   = wave - 256;
            float s = 0.f;
            for (int k = lane; k < Hsz; k += 32) s += h2[b * Hsz + k] * Wout[k];
            for (int off = 16; off > 0; off >>= 1) s += __shfl_down(s, off, 32);
            if (lane == 0) y[b * Tsz + (t - 1)] = s + bout[0];
        }
        gridbar(ctrl);

        // ---- phase C: GI2 = h1b * Wi2^T; WG pairs share a strip, split by M
        {
            int mb = ((blockIdx.x & 1) * 8 + wiw) * 16;   // one mtile per wave
            int nb = (blockIdx.x >> 1) * SCOLS;
            gemm_strip_lds<1>(h1b, sWi, GI2, mb, nb, lane);
        }
        gridbar(ctrl);

        // ---- phase D: h2 gate update, all 16384 threads, 8 elems each
#pragma unroll 4
        for (int s = 0; s < 8; ++s) {
            int e = gtid + s * 16384;
            int b = e >> 9, j = e & 511;
            float gr = GI2[(size_t)b * G3 + j]        + bih2[j]        + GH2[(size_t)b * G3 + j]        + bhh2[j];
            float gz = GI2[(size_t)b * G3 + j + 512]  + bih2[j + 512]  + GH2[(size_t)b * G3 + j + 512]  + bhh2[j + 512];
            float gn = GI2[(size_t)b * G3 + j + 1024] + bih2[j + 1024];
            float hn = GH2[(size_t)b * G3 + j + 1024] + bhh2[j + 1024];
            float r = sigm(gr), z = sigm(gz);
            float n  = tanhf(gn + r * hn);
            float hv = (1.0f - z) * n + z * h2[e];
            h2[e]  = hv;
            h2b[e] = f2bf(hv);
        }
        gridbar(ctrl);
    }

    // ---- final readout y[:, T-1]
    if (wave >= 256) {
        int b   = wave - 256;
        float s = 0.f;
        for (int k = lane; k < Hsz; k += 32) s += h2[b * Hsz + k] * Wout[k];
        for (int off = 16; off > 0; off >>= 1) s += __shfl_down(s, off, 32);
        if (lane == 0) y[b * Tsz + (Tsz - 1)] = s + bout[0];
    }
}

// ---------------------------------------------------------------------------
extern "C" void kernel_launch(void* const* d_in, const int* in_sizes, int n_in,
                              void* d_out, int out_size, void* d_ws, size_t ws_size,
                              hipStream_t stream) {
    const float* x    = (const float*)d_in[0];
    const float* Wih1 = (const float*)d_in[1];
    const float* Whh1 = (const float*)d_in[2];
    const float* bih1 = (const float*)d_in[3];
    const float* bhh1 = (const float*)d_in[4];
    const float* Wih2 = (const float*)d_in[5];
    const float* Whh2 = (const float*)d_in[6];
    const float* bih2 = (const float*)d_in[7];
    const float* bhh2 = (const float*)d_in[8];
    const float* Wout = (const float*)d_in[9];
    const float* bout = (const float*)d_in[10];
    float* y = (float*)d_out;

    char*  base = (char*)d_ws;
    size_t off  = 0;
    auto alloc = [&](size_t bytes) -> void* {
        off = (off + 255) & ~(size_t)255;
        void* r = base + off;
        off += bytes;
        return r;
    };
    int*            ctrl = (int*)alloc(2 * sizeof(int));
    unsigned short* wc1  = (unsigned short*)alloc((size_t)G3 * Hsz * 2);
    unsigned short* wi2  = (unsigned short*)alloc((size_t)G3 * Hsz * 2);
    unsigned short* wh2  = (unsigned short*)alloc((size_t)G3 * Hsz * 2);
    float*          h1   = (float*)alloc((size_t)Bsz * Hsz * 4);
    float*          h2   = (float*)alloc((size_t)Bsz * Hsz * 4);
    unsigned short* h1b  = (unsigned short*)alloc((size_t)Bsz * Hsz * 2);
    unsigned short* h2b  = (unsigned short*)alloc((size_t)Bsz * Hsz * 2);
    float*          GH1  = (float*)alloc((size_t)Bsz * G3 * 4);
    float*          GI2  = (float*)alloc((size_t)Bsz * G3 * 4);
    float*          GH2  = (float*)alloc((size_t)Bsz * G3 * 4);

    gru_prep<<<dim3(768), dim3(256), 0, stream>>>(Whh1, Wih2, Whh2, wc1, wi2, wh2,
                                                  h1, h2, h1b, h2b, ctrl);
    gru_persistent<<<dim3(NWG), dim3(TPB), SMEM_BYTES, stream>>>(
        x, Wih1, bih1, bhh1, bih2, bhh2, Wout, bout, y,
        wc1, wi2, wh2, h1, h2, h1b, h2b, GH1, GI2, GH2, ctrl);
}
